// SDFDecoder_75041668596212
// MI455X (gfx1250) — matrix operand order repaired and checked by reading, not compiled, not run on hardware
//
#include <hip/hip_runtime.h>
#include <hip/hip_bf16.h>
#include <math.h>

// ---------------------------------------------------------------------------
// SDF decoder for MI455X (gfx1250, wave32, WMMA).
// 64 points per block (4 row-tiles). MLP as 16x16x32 f16 WMMA. Weights packed
// per-launch into B-fragment lane layout. 512-wide layers: each wave processes
// a PAIR of col-tiles with 8 accumulators, so each A fragment (2x ds_load_b128)
// feeds 2 WMMAs -> 1 LDS instruction per WMMA. Softplus is branchless exp/log.
// ---------------------------------------------------------------------------

typedef __attribute__((ext_vector_type(16))) _Float16 v16h;
typedef __attribute__((ext_vector_type(8)))  float    v8f;
typedef __attribute__((ext_vector_type(4)))  uint32_t v4u;

#define BB 2
#define MM 4
#define CC 32
#define DD 64
#define HH 128
#define WW 128
#define NN 65536

#define PTS_PER_BLK 64
#define S0 152   // LDS stride (halfs), 16B-aligned rows; 76 words % 64 = 12 -> 16 distinct banks
#define SA 536   // LDS stride (halfs), 16B-aligned rows; 268 words % 64 = 12

// packed-weight offsets in halfs inside workspace
#define OFF1 0
#define OFF2 (OFF1 + 3*32*512)       // W1: 96x512  -> 3 ktiles  x 32 ntiles
#define OFF3 (OFF2 + 16*32*512)      // W2: 512x512 -> 16 ktiles x 32 ntiles
#define OFF4 (OFF3 + 16*32*512)      // W3: 512x512
#define PACK_TOTAL (OFF4 + 16*4*512) // + W4: 512x64 -> 16 ktiles x 4 ntiles

// ---------------------------------------------------------------------------
// Pack f32 row-major W[K][N] into f16 WMMA B fragments.
//   idx = ((kt*ntiles + nt)*32 + lane)*16 + i
//   lane 0..15 : K = kt*32 + (i&7) + 16*(i>>3),      N = nt*16 + lane
//   lane 16..31: K = kt*32 + 8 + (i&7) + 16*(i>>3),  N = nt*16 + (lane-16)
// ---------------------------------------------------------------------------
__global__ __launch_bounds__(256) void pack_w_kernel(const float* __restrict__ W,
                                                     _Float16* __restrict__ dst,
                                                     int K, int Nout, int ntiles) {
  int idx  = blockIdx.x * 256 + threadIdx.x;
  int i    = idx & 15;
  int lane = (idx >> 4) & 31;
  int tile = idx >> 9;
  int kt   = tile / ntiles;
  int nt   = tile - kt * ntiles;
  int k = kt * 32 + ((lane >> 4) << 3) + (i & 7) + ((i >> 3) << 4);
  int n = nt * 16 + (lane & 15);
  float w = (k < K && n < Nout) ? W[(size_t)k * Nout + n] : 0.0f;
  dst[idx] = (_Float16)w;
}

// softplus(100v)/100, branchless: max(z,0) + log1p(exp(-|z|)) via fast HW ops
__device__ __forceinline__ float sp100(float v) {
  float z = 100.0f * v;
  float e = __expf(-fabsf(z));
  return (fmaxf(z, 0.0f) + __logf(1.0f + e)) * 0.01f;
}

// A-fragment (16x32 f16) from LDS row-major activations.
// Per lane the fragment is two contiguous 16-byte chunks:
//   halfs [kbase, kbase+8) and [kbase+16, kbase+24), kbase = kt*32 + 8*(lane>=16)
// -> 2x ds_load_b128 (rows are 16B aligned since SA/S0 are multiples of 8).
__device__ __forceinline__ v16h load_a_frag(const _Float16* src, int stride, int kt, int lane) {
  int M     = lane & 15;
  int kbase = kt * 32 + ((lane >> 4) << 3);
  const _Float16* p = src + M * stride + kbase;
  union { v16h v; v4u q[2]; } a;
  a.q[0] = *reinterpret_cast<const v4u*>(p);
  a.q[1] = *reinterpret_cast<const v4u*>(p + 16);
  return a.v;
}

#define WMMA_F16(a, b, c) \
  __builtin_amdgcn_wmma_f32_16x16x32_f16(false, (a), false, (b), (short)0, (c), false, false)

// Dense layer (512 outputs, ntiles==32): wave processes col-tile pair
// (nt, nt+8) with 8 accumulators; each A fragment feeds both B tiles.
__device__ __forceinline__ void dense_layer_pair(const _Float16* src, int sstride, int ktiles,
                                                 _Float16* dst, int dstride,
                                                 const _Float16* __restrict__ Wp,
                                                 const float* __restrict__ bias,
                                                 int wave, int lane) {
#pragma unroll 1
  for (int nt0 = wave; nt0 < 32; nt0 += 16) {
    v8f a0a = {}, a1a = {}, a2a = {}, a3a = {};
    v8f a0b = {}, a1b = {}, a2b = {}, a3b = {};
    const _Float16* wA = Wp + ((size_t)nt0 * 32 + lane) * 16;
    const _Float16* wB = Wp + ((size_t)(nt0 + 8) * 32 + lane) * 16;
#pragma unroll 1
    for (int kt = 0; kt < ktiles; ++kt) {
      v16h bA = *reinterpret_cast<const v16h*>(wA + (size_t)kt * 32 * 512);
      v16h bB = *reinterpret_cast<const v16h*>(wB + (size_t)kt * 32 * 512);
      v16h a0 = load_a_frag(src,                sstride, kt, lane);
      a0a = WMMA_F16(a0, bA, a0a);  a0b = WMMA_F16(a0, bB, a0b);
      v16h a1 = load_a_frag(src + 16 * sstride, sstride, kt, lane);
      a1a = WMMA_F16(a1, bA, a1a);  a1b = WMMA_F16(a1, bB, a1b);
      v16h a2 = load_a_frag(src + 32 * sstride, sstride, kt, lane);
      a2a = WMMA_F16(a2, bA, a2a);  a2b = WMMA_F16(a2, bB, a2b);
      v16h a3 = load_a_frag(src + 48 * sstride, sstride, kt, lane);
      a3a = WMMA_F16(a3, bA, a3a);  a3b = WMMA_F16(a3, bB, a3b);
    }
    int nca = nt0 * 16 + (lane & 15);
    int ncb = (nt0 + 8) * 16 + (lane & 15);
    float bva = bias[nca], bvb = bias[ncb];
    int mbase = (lane >> 4) << 3;           // lanes 16-31 hold rows M=8..15
    _Float16* da = dst + (size_t)mbase * dstride + nca;
    _Float16* db = dst + (size_t)mbase * dstride + ncb;
#pragma unroll
    for (int r = 0; r < 8; ++r) da[(size_t)r * dstride]        = (_Float16)sp100(a0a[r] + bva);
#pragma unroll
    for (int r = 0; r < 8; ++r) da[(size_t)(16 + r) * dstride] = (_Float16)sp100(a1a[r] + bva);
#pragma unroll
    for (int r = 0; r < 8; ++r) da[(size_t)(32 + r) * dstride] = (_Float16)sp100(a2a[r] + bva);
#pragma unroll
    for (int r = 0; r < 8; ++r) da[(size_t)(48 + r) * dstride] = (_Float16)sp100(a3a[r] + bva);
#pragma unroll
    for (int r = 0; r < 8; ++r) db[(size_t)r * dstride]        = (_Float16)sp100(a0b[r] + bvb);
#pragma unroll
    for (int r = 0; r < 8; ++r) db[(size_t)(16 + r) * dstride] = (_Float16)sp100(a1b[r] + bvb);
#pragma unroll
    for (int r = 0; r < 8; ++r) db[(size_t)(32 + r) * dstride] = (_Float16)sp100(a2b[r] + bvb);
#pragma unroll
    for (int r = 0; r < 8; ++r) db[(size_t)(48 + r) * dstride] = (_Float16)sp100(a3b[r] + bvb);
  }
}

// Dense layer, generic single-tile variant (used for the 64-wide L4).
__device__ __forceinline__ void dense_layer_single(const _Float16* src, int sstride, int ktiles,
                                                   _Float16* dst, int dstride, int ntiles,
                                                   const _Float16* __restrict__ Wp,
                                                   const float* __restrict__ bias,
                                                   int wave, int lane) {
#pragma unroll 1
  for (int nt = wave; nt < ntiles; nt += 8) {
    v8f acc0 = {}, acc1 = {}, acc2 = {}, acc3 = {};
    const _Float16* wbase = Wp + ((size_t)nt * 32 + lane) * 16;
#pragma unroll 1
    for (int kt = 0; kt < ktiles; ++kt) {
      v16h b = *reinterpret_cast<const v16h*>(wbase + (size_t)kt * ntiles * 512);
      v16h a0 = load_a_frag(src,                sstride, kt, lane);
      acc0 = WMMA_F16(a0, b, acc0);
      v16h a1 = load_a_frag(src + 16 * sstride, sstride, kt, lane);
      acc1 = WMMA_F16(a1, b, acc1);
      v16h a2 = load_a_frag(src + 32 * sstride, sstride, kt, lane);
      acc2 = WMMA_F16(a2, b, acc2);
      v16h a3 = load_a_frag(src + 48 * sstride, sstride, kt, lane);
      acc3 = WMMA_F16(a3, b, acc3);
    }
    int ncol  = nt * 16 + (lane & 15);
    float bv  = bias[ncol];
    int mbase = (lane >> 4) << 3;
    _Float16* drow = dst + (size_t)mbase * dstride + ncol;
#pragma unroll
    for (int r = 0; r < 8; ++r) drow[(size_t)r * dstride]        = (_Float16)sp100(acc0[r] + bv);
#pragma unroll
    for (int r = 0; r < 8; ++r) drow[(size_t)(16 + r) * dstride] = (_Float16)sp100(acc1[r] + bv);
#pragma unroll
    for (int r = 0; r < 8; ++r) drow[(size_t)(32 + r) * dstride] = (_Float16)sp100(acc2[r] + bv);
#pragma unroll
    for (int r = 0; r < 8; ++r) drow[(size_t)(48 + r) * dstride] = (_Float16)sp100(acc3[r] + bv);
  }
}

__global__ __launch_bounds__(256)
__attribute__((amdgpu_waves_per_eu(4)))
void sdf_main_kernel(
    const float* __restrict__ pts,     const float* __restrict__ volume,
    const float* __restrict__ corners, const float* __restrict__ proj,
    const float* __restrict__ fmap,
    const float* __restrict__ b1, const float* __restrict__ b2,
    const float* __restrict__ b3, const float* __restrict__ b4,
    const float* __restrict__ W5, const float* __restrict__ b5,
    const _Float16* __restrict__ packed, float* __restrict__ out) {
  __shared__ __align__(16) _Float16 x0[PTS_PER_BLK * S0];   // ~19.5 KB
  __shared__ __align__(16) _Float16 xa[PTS_PER_BLK * SA];   // ~68.6 KB
  __shared__ __align__(16) _Float16 xb[PTS_PER_BLK * SA];   // ~68.6 KB

  const int tid  = threadIdx.x;
  const int lane = tid & 31;
  const int wave = tid >> 5;

  // ---- feature gather: 16 threads/point, 2 channels/thread, 4 point groups --
#pragma unroll 1
  for (int pg = 0; pg < 4; ++pg) {
    const int p_local = pg * 16 + (tid >> 4);
    const int cslot   = tid & 15;
    const long long gp = (long long)blockIdx.x * PTS_PER_BLK + p_local;
    const int b = (int)(gp >> 16);
    const int n = (int)(gp & (NN - 1));

    const float px = pts[((size_t)b * NN + n) * 3 + 0];
    const float py = pts[((size_t)b * NN + n) * 3 + 1];
    const float pz = pts[((size_t)b * NN + n) * 3 + 2];

    // trilinear volume sample -> channels [0,32)
    {
      const float lo0 = corners[b * 6 + 0], lo1 = corners[b * 6 + 1], lo2 = corners[b * 6 + 2];
      const float hi0 = corners[b * 6 + 3], hi1 = corners[b * 6 + 4], hi2 = corners[b * 6 + 5];
      float gx = (px - lo0) / (hi0 - lo0) * 2.0f - 1.0f;
      float gy = (py - lo1) / (hi1 - lo1) * 2.0f - 1.0f;
      float gz = (pz - lo2) / (hi2 - lo2) * 2.0f - 1.0f;
      float ix = (gx + 1.0f) * 0.5f * (float)(DD - 1);
      float iy = (gy + 1.0f) * 0.5f * (float)(DD - 1);
      float iz = (gz + 1.0f) * 0.5f * (float)(DD - 1);
      float xf = floorf(ix), yf = floorf(iy), zf = floorf(iz);
      float wx = ix - xf, wy = iy - yf, wz = iz - zf;
      int xi0 = (int)fminf(fmaxf(xf,        0.0f), 63.0f);
      int xi1 = (int)fminf(fmaxf(xf + 1.0f, 0.0f), 63.0f);
      int yi0 = (int)fminf(fmaxf(yf,        0.0f), 63.0f);
      int yi1 = (int)fminf(fmaxf(yf + 1.0f, 0.0f), 63.0f);
      int zi0 = (int)fminf(fmaxf(zf,        0.0f), 63.0f);
      int zi1 = (int)fminf(fmaxf(zf + 1.0f, 0.0f), 63.0f);
      const size_t vb = (size_t)b * CC * (DD * DD * DD);
#pragma unroll
      for (int cc = 0; cc < 2; ++cc) {
        const int ch = cslot + cc * 16;
        const float* V = volume + vb + (size_t)ch * (DD * DD * DD);
        float c000 = V[((size_t)zi0 * DD + yi0) * DD + xi0];
        float c001 = V[((size_t)zi0 * DD + yi0) * DD + xi1];
        float c010 = V[((size_t)zi0 * DD + yi1) * DD + xi0];
        float c011 = V[((size_t)zi0 * DD + yi1) * DD + xi1];
        float c100 = V[((size_t)zi1 * DD + yi0) * DD + xi0];
        float c101 = V[((size_t)zi1 * DD + yi0) * DD + xi1];
        float c110 = V[((size_t)zi1 * DD + yi1) * DD + xi0];
        float c111 = V[((size_t)zi1 * DD + yi1) * DD + xi1];
        float r = c000 * (1 - wx) * (1 - wy) * (1 - wz) + c001 * wx * (1 - wy) * (1 - wz)
                + c010 * (1 - wx) * wy * (1 - wz)       + c011 * wx * wy * (1 - wz)
                + c100 * (1 - wx) * (1 - wy) * wz       + c101 * wx * (1 - wy) * wz
                + c110 * (1 - wx) * wy * wz             + c111 * wx * wy * wz;
        x0[p_local * S0 + ch] = (_Float16)r;
      }
    }

    // bilinear camera features, mean over M -> channels [32,64)
    {
      float acc0 = 0.0f, acc1 = 0.0f;
#pragma unroll 1
      for (int m = 0; m < MM; ++m) {
        const float* P = proj + ((size_t)b * MM + m) * 16;
        float u0 = P[0] * px + P[1] * py + P[2]  * pz + P[3];
        float v0 = P[4] * px + P[5] * py + P[6]  * pz + P[7];
        float z0 = P[8] * px + P[9] * py + P[10] * pz + P[11];
        float den = fmaxf(z0, 1e-16f);
        float u = (u0 / den) / (float)(WW - 1) * 2.0f - 1.0f;
        float v = (v0 / den) / (float)(HH - 1) * 2.0f - 1.0f;
        if (fabsf(u) > 1.0f) u = 2.0f;
        if (fabsf(v) > 1.0f) v = 2.0f;
        float ix = (u + 1.0f) * 0.5f * (float)(WW - 1);
        float iy = (v + 1.0f) * 0.5f * (float)(HH - 1);
        float xf = floorf(ix), yf = floorf(iy);
        float wx = ix - xf, wy = iy - yf;
        int xi0 = (int)fminf(fmaxf(xf,        0.0f), 127.0f);
        int xi1 = (int)fminf(fmaxf(xf + 1.0f, 0.0f), 127.0f);
        int yi0 = (int)fminf(fmaxf(yf,        0.0f), 127.0f);
        int yi1 = (int)fminf(fmaxf(yf + 1.0f, 0.0f), 127.0f);
        float w00 = (1 - wx) * (1 - wy), w10 = wx * (1 - wy);
        float w01 = (1 - wx) * wy,       w11 = wx * wy;
        const size_t fb = ((size_t)b * MM + m) * CC * (HH * WW);
        const float* F0 = fmap + fb + (size_t)cslot        * (HH * WW);
        const float* F1 = fmap + fb + (size_t)(cslot + 16) * (HH * WW);
        acc0 += F0[yi0 * WW + xi0] * w00 + F0[yi0 * WW + xi1] * w10
              + F0[yi1 * WW + xi0] * w01 + F0[yi1 * WW + xi1] * w11;
        acc1 += F1[yi0 * WW + xi0] * w00 + F1[yi0 * WW + xi1] * w10
              + F1[yi1 * WW + xi0] * w01 + F1[yi1 * WW + xi1] * w11;
      }
      x0[p_local * S0 + 32 + cslot] = (_Float16)(acc0 * 0.25f);
      x0[p_local * S0 + 48 + cslot] = (_Float16)(acc1 * 0.25f);
    }

    // positional embedding -> channels [64,79), zero-pad [79,96)
    if (cslot < 15) {
      int grp = cslot / 3, ax = cslot - grp * 3;
      float val = (ax == 0) ? px : ((ax == 1) ? py : pz);
      float e;
      if      (grp == 0) e = val;
      else if (grp == 1) e = sinf(val);
      else if (grp == 2) e = cosf(val);
      else if (grp == 3) e = sinf(2.0f * val);
      else               e = cosf(2.0f * val);
      x0[p_local * S0 + 64 + cslot] = (_Float16)e;
    } else {
      for (int k = 79; k < 96; ++k) x0[p_local * S0 + k] = (_Float16)0.0f;
    }
  }

  __syncthreads();

  // ---- MLP via WMMA ----
  dense_layer_pair(x0, S0, 3,  xa, SA, packed + OFF1, b1, wave, lane);
  __syncthreads();
  dense_layer_pair(xa, SA, 16, xb, SA, packed + OFF2, b2, wave, lane);
  __syncthreads();
  dense_layer_pair(xb, SA, 16, xa, SA, packed + OFF3, b3, wave, lane);
  __syncthreads();
  dense_layer_single(xa, SA, 16, xb, SA, 4, packed + OFF4, b4, wave, lane);
  __syncthreads();

  // final 64 -> 1 dot product, one thread per point
  if (tid < PTS_PER_BLK) {
    float s = b5[0];
#pragma unroll 8
    for (int k = 0; k < 64; ++k) s += (float)xb[tid * SA + k] * W5[k];
    out[(long long)blockIdx.x * PTS_PER_BLK + tid] = s;
  }
}

extern "C" void kernel_launch(void* const* d_in, const int* in_sizes, int n_in,
                              void* d_out, int out_size, void* d_ws, size_t ws_size,
                              hipStream_t stream) {
  const float* pts     = (const float*)d_in[0];
  const float* volume  = (const float*)d_in[1];
  const float* corners = (const float*)d_in[2];
  const float* proj    = (const float*)d_in[3];
  const float* fmap    = (const float*)d_in[4];
  const float* W1 = (const float*)d_in[5];  const float* b1 = (const float*)d_in[6];
  const float* W2 = (const float*)d_in[7];  const float* b2 = (const float*)d_in[8];
  const float* W3 = (const float*)d_in[9];  const float* b3 = (const float*)d_in[10];
  const float* W4 = (const float*)d_in[11]; const float* b4 = (const float*)d_in[12];
  const float* W5 = (const float*)d_in[13]; const float* b5 = (const float*)d_in[14];
  float* out = (float*)d_out;
  _Float16* packed = (_Float16*)d_ws;   // needs PACK_TOTAL*2 = ~1.2 MB

  // pack weights into WMMA B-fragment layout (runs every call; deterministic)
  pack_w_kernel<<<(3  * 32 * 512) / 256, 256, 0, stream>>>(W1, packed + OFF1,  79, 512, 32);
  pack_w_kernel<<<(16 * 32 * 512) / 256, 256, 0, stream>>>(W2, packed + OFF2, 512, 512, 32);
  pack_w_kernel<<<(16 * 32 * 512) / 256, 256, 0, stream>>>(W3, packed + OFF3, 512, 512, 32);
  pack_w_kernel<<<(16 * 4  * 512) / 256, 256, 0, stream>>>(W4, packed + OFF4, 512,  64,  4);

  // 64 points per block over B*N points
  const int nblocks = (BB * NN) / PTS_PER_BLK;   // 2048
  sdf_main_kernel<<<nblocks, 256, 0, stream>>>(pts, volume, corners, proj, fmap,
                                               b1, b2, b3, b4, W5, b5, packed, out);
}
